// SimpleDecoder_43757126812132
// MI455X (gfx1250) — matrix operand order repaired and checked by reading, not compiled, not run on hardware
//
#include <hip/hip_runtime.h>
#include <hip/hip_fp16.h>

// ---------------------------------------------------------------------------
// MI455X (gfx1250) transformer decoder block, f16-WMMA everywhere.
// D=1024, H=16, HD=64, S=2048, B=2.  All GEMMs and attention use
// v_wmma_f32_16x16x32_f16 (f16 inputs, f32 accumulate).
// GEMM staging: GLOBAL_LOAD_ASYNC_TO_LDS_B128 with double-buffered LDS and
// s_wait_asynccnt pipelining (builtin confirmed present; signature takes
// typed v4i pointers per the round-2 diagnostic).
// ---------------------------------------------------------------------------

typedef __attribute__((ext_vector_type(16))) _Float16 v16h;
typedef __attribute__((ext_vector_type(8)))  _Float16 v8h;
typedef __attribute__((ext_vector_type(8)))  float    v8f;
typedef int v4i_vs __attribute__((vector_size(16)));

#define D_MODEL 1024
#define NHEAD   16
#define HDIM    64
#define SEQ     2048
#define BATCH   2
#define MROWS   (BATCH * SEQ)     // 4096
#define DFF     (4 * D_MODEL)     // 4096
#define NEGINF  (-1e20f)

#if __has_builtin(__builtin_amdgcn_global_load_async_to_lds_b128)
#define HAVE_ASYNC_LDS 1
#else
#define HAVE_ASYNC_LDS 0
#endif

__device__ __forceinline__ void cp_async16(_Float16* lds, const _Float16* g) {
#if HAVE_ASYNC_LDS
  __builtin_amdgcn_global_load_async_to_lds_b128(
      (__attribute__((address_space(1))) v4i_vs*)(v4i_vs*)(void*)const_cast<_Float16*>(g),
      (__attribute__((address_space(3))) v4i_vs*)(v4i_vs*)(void*)lds,
      /*offset=*/0, /*cpol=*/0);
#else
  (void)lds; (void)g;
#endif
}

template <int N>
__device__ __forceinline__ void wait_async() {
#if __has_builtin(__builtin_amdgcn_s_wait_asynccnt)
  __builtin_amdgcn_s_wait_asynccnt(N);
#else
  asm volatile("s_wait_asynccnt %0" :: "i"(N) : "memory");
#endif
}

__device__ __forceinline__ v8f wmma_f16(v16h a, v16h b, v8f c) {
  // D = A(16x32 f16) * B(32x16 f16) + C(16x16 f32)
  return __builtin_amdgcn_wmma_f32_16x16x32_f16(
      /*neg_a=*/false, a, /*neg_b=*/false, b,
      /*c_mod=*/(short)0, c, /*reuse_a=*/false, /*reuse_b=*/false);
}

// A-fragment (16x32, K contiguous in memory).  ISA layout: lane<16 holds row
// M=lane, K={0..7,16..23}; lane>=16 holds M=lane-16, K={8..15,24..31}.
__device__ __forceinline__ v16h frag_a(const _Float16* __restrict__ t, int pitch, int lane) {
  const int m  = lane & 15;
  const int ko = (lane >> 4) << 3;          // 0 or 8
  v8h lo = *(const v8h*)(t + (size_t)m * pitch + ko);
  v8h hi = *(const v8h*)(t + (size_t)m * pitch + ko + 16);
  v16h f;
#pragma unroll
  for (int i = 0; i < 8; ++i) { f[i] = lo[i]; f[i + 8] = hi[i]; }
  return f;
}

// B-fragment (32x16).  ISA layout: lanes 0-15 hold K={0..7,16..23} for N=lane, lanes
// 16-31 hold K={8..15,24..31}.  Source tile is stored [N][K] with K contiguous.
__device__ __forceinline__ v16h frag_b(const _Float16* __restrict__ t, int pitch, int lane) {
  const int n  = lane & 15;
  const int ko = (lane >> 4) << 3;          // 0 or 8
  v8h lo = *(const v8h*)(t + (size_t)n * pitch + ko);
  v8h hi = *(const v8h*)(t + (size_t)n * pitch + ko + 16);
  v16h f;
#pragma unroll
  for (int i = 0; i < 8; ++i) { f[i] = lo[i]; f[i + 8] = hi[i]; }
  return f;
}

// ---------------------------------------------------------------------------
// fp32 -> f16 conversion
// ---------------------------------------------------------------------------
__global__ void cvt_f16_kernel(const float* __restrict__ in,
                               _Float16* __restrict__ out, int n) {
  int i = blockIdx.x * blockDim.x + threadIdx.x;
  if (i < n) out[i] = (_Float16)in[i];
}

// ---------------------------------------------------------------------------
// C[M,N] = A[M,K] * Bw[N,K]^T + bias  (both A and Bw are K-contiguous).
// 64x64 tile per 128-thread block (4 waves, 2x2 of 32x32 per-wave tiles),
// K-step 32.  Async-to-LDS double-buffered pipeline.
// ---------------------------------------------------------------------------
#define GT 64
#define GK 32
#define AP 40    // LDS row pitch in halves (80B, 16B-aligned rows, depads banks)

__global__ __launch_bounds__(128)
void gemm_kernel(const _Float16* __restrict__ A, const _Float16* __restrict__ Bw,
                 const float* __restrict__ bias, float* __restrict__ Cf,
                 _Float16* __restrict__ Ch, int M, int N, int K, int relu) {
  __shared__ alignas(16) _Float16 As[2][GT * AP];
  __shared__ alignas(16) _Float16 Bs[2][GT * AP];
  const int tid  = threadIdx.x;
  const int lane = tid & 31;
  const int wave = tid >> 5;
  const int m0 = blockIdx.x * GT;
  const int n0 = blockIdx.y * GT;
  const int wm = (wave >> 1) * 32;
  const int wn = (wave & 1) * 32;

  v8f acc[2][2] = {};
  const int r = tid >> 1;           // 0..63: tile row staged by this thread
  const int c = (tid & 1) * 16;     // half-row
  const _Float16* ga = A  + (size_t)(m0 + r) * K + c;
  const _Float16* gb = Bw + (size_t)(n0 + r) * K + c;

#if HAVE_ASYNC_LDS
  // ---- async double-buffered pipeline: 4 async b128 copies per thread/tile
  auto stage = [&](int buf, int k0) {
    cp_async16(&As[buf][r * AP + c],     ga + k0);
    cp_async16(&As[buf][r * AP + c + 8], ga + k0 + 8);
    cp_async16(&Bs[buf][r * AP + c],     gb + k0);
    cp_async16(&Bs[buf][r * AP + c + 8], gb + k0 + 8);
  };
  const int nk = K / GK;
  stage(0, 0);
  for (int i = 0; i < nk; ++i) {
    const int buf = i & 1;
    if (i + 1 < nk) {
      stage(buf ^ 1, (i + 1) * GK);   // overlap next tile copy with compute
      wait_async<4>();                // in-order: tile i fully resident
    } else {
      wait_async<0>();
    }
    __syncthreads();                  // tile i visible to all waves
    v16h af0 = frag_a(As[buf] + (wm)      * AP, AP, lane);
    v16h af1 = frag_a(As[buf] + (wm + 16) * AP, AP, lane);
    v16h bf0 = frag_b(Bs[buf] + (wn)      * AP, AP, lane);
    v16h bf1 = frag_b(Bs[buf] + (wn + 16) * AP, AP, lane);
    acc[0][0] = wmma_f16(af0, bf0, acc[0][0]);
    acc[0][1] = wmma_f16(af0, bf1, acc[0][1]);
    acc[1][0] = wmma_f16(af1, bf0, acc[1][0]);
    acc[1][1] = wmma_f16(af1, bf1, acc[1][1]);
    __syncthreads();                  // all reads done before buf is re-filled
  }
#else
  // ---- fallback: synchronous VGPR staging (single buffer)
  for (int k0 = 0; k0 < K; k0 += GK) {
    v8h a0v = *(const v8h*)(ga + k0);
    v8h a1v = *(const v8h*)(ga + k0 + 8);
    v8h b0v = *(const v8h*)(gb + k0);
    v8h b1v = *(const v8h*)(gb + k0 + 8);
    *(v8h*)(&As[0][r * AP + c])     = a0v;
    *(v8h*)(&As[0][r * AP + c + 8]) = a1v;
    *(v8h*)(&Bs[0][r * AP + c])     = b0v;
    *(v8h*)(&Bs[0][r * AP + c + 8]) = b1v;
    __syncthreads();
    v16h af0 = frag_a(As[0] + (wm)      * AP, AP, lane);
    v16h af1 = frag_a(As[0] + (wm + 16) * AP, AP, lane);
    v16h bf0 = frag_b(Bs[0] + (wn)      * AP, AP, lane);
    v16h bf1 = frag_b(Bs[0] + (wn + 16) * AP, AP, lane);
    acc[0][0] = wmma_f16(af0, bf0, acc[0][0]);
    acc[0][1] = wmma_f16(af0, bf1, acc[0][1]);
    acc[1][0] = wmma_f16(af1, bf0, acc[1][0]);
    acc[1][1] = wmma_f16(af1, bf1, acc[1][1]);
    __syncthreads();
  }
#endif

  // C layout: lane holds N=lane%16; VGPR rr -> M = rr + 8*(lane/16).
  const int n    = lane & 15;
  const int rsel = (lane >> 4) * 8;
#pragma unroll
  for (int i = 0; i < 2; ++i)
#pragma unroll
    for (int j = 0; j < 2; ++j) {
      const int gn = n0 + wn + j * 16 + n;
      const float bv = bias[gn];
#pragma unroll
      for (int rr = 0; rr < 8; ++rr) {
        const int gm = m0 + wm + i * 16 + rsel + rr;
        float v = acc[i][j][rr] + bv;
        if (relu) v = fmaxf(v, 0.0f);
        if (Cf) Cf[(size_t)gm * N + gn] = v;
        if (Ch) Ch[(size_t)gm * N + gn] = (_Float16)v;
      }
    }
}

// ---------------------------------------------------------------------------
// Causal flash attention.  Grid: (S/64, H, B); 128 threads; each wave owns a
// 16-row q-tile.  K tile [32 x 64] staged row-major; V tile staged transposed
// so the PV B-fragment is a contiguous LDS load.
// ---------------------------------------------------------------------------
#define KS_P 72
#define VT_P 40
#define PS_P 40

__global__ __launch_bounds__(128)
void attn_kernel(const _Float16* __restrict__ Q, const _Float16* __restrict__ Km,
                 const _Float16* __restrict__ V, _Float16* __restrict__ O) {
  __shared__ alignas(16) _Float16 Ks[32 * KS_P];
  __shared__ alignas(16) _Float16 Vt[64 * VT_P];
  __shared__ alignas(16) _Float16 Ps[4][16 * PS_P];

  const int tid  = threadIdx.x;
  const int lane = tid & 31;
  const int wave = tid >> 5;
  const int qblk = blockIdx.x;
  const int h    = blockIdx.y;
  const int b    = blockIdx.z;
  const size_t rowbase = (size_t)b * SEQ;
  const int hoff = h * HDIM;

  const int qbase = qblk * 64 + wave * 16;
  const int q_hi  = qbase + 15;
  const int kmax  = qblk * 64 + 64;

  const _Float16* Qp = Q  + (rowbase + qbase) * D_MODEL + hoff;
  const _Float16* Kp = Km + rowbase * D_MODEL + hoff;
  const _Float16* Vp = V  + rowbase * D_MODEL + hoff;

  // Q fragments (16 x 64 head slice), loaded straight from global.
  const v16h aq0 = frag_a(Qp,      D_MODEL, lane);
  const v16h aq1 = frag_a(Qp + 32, D_MODEL, lane);

  float rmax[8], rsum[8];
  v8f o[4] = {};
#pragma unroll
  for (int r = 0; r < 8; ++r) { rmax[r] = -__builtin_inff(); rsum[r] = 0.0f; }

  const int n    = lane & 15;
  const int rsel = (lane >> 4) * 8;

  for (int k0 = 0; k0 < kmax; k0 += 32) {
    { // stage K row-major and V transposed
      const int kr = tid >> 2;          // 0..31
      const int dc = (tid & 3) * 16;    // 0,16,32,48
      const _Float16* kg = Kp + (size_t)(k0 + kr) * D_MODEL + dc;
      const _Float16* vg = Vp + (size_t)(k0 + kr) * D_MODEL + dc;
      v8h k0v = *(const v8h*)(kg);
      v8h k1v = *(const v8h*)(kg + 8);
      *(v8h*)(Ks + kr * KS_P + dc)     = k0v;
      *(v8h*)(Ks + kr * KS_P + dc + 8) = k1v;
      v8h v0v = *(const v8h*)(vg);
      v8h v1v = *(const v8h*)(vg + 8);
#pragma unroll
      for (int i = 0; i < 8; ++i) {
        Vt[(dc + i)     * VT_P + kr] = v0v[i];
        Vt[(dc + 8 + i) * VT_P + kr] = v1v[i];
      }
    }
    __syncthreads();

    if (k0 <= q_hi) {                  // wave-uniform: EXEC stays all-ones
      v8f s0 = {}, s1 = {};
      {
        v16h bk00 = frag_b(Ks,                   KS_P, lane); // keys 0-15,  d 0-31
        v16h bk01 = frag_b(Ks + 32,              KS_P, lane); // keys 0-15,  d 32-63
        v16h bk10 = frag_b(Ks + 16 * KS_P,       KS_P, lane); // keys 16-31, d 0-31
        v16h bk11 = frag_b(Ks + 16 * KS_P + 32,  KS_P, lane);
        s0 = wmma_f16(aq0, bk00, s0);
        s0 = wmma_f16(aq1, bk01, s0);
        s1 = wmma_f16(aq0, bk10, s1);
        s1 = wmma_f16(aq1, bk11, s1);
      }
#pragma unroll
      for (int r = 0; r < 8; ++r) {
        const int q = qbase + rsel + r;
        float x0 = s0[r] * 0.125f;                 // 1/sqrt(64)
        float x1 = s1[r] * 0.125f;
        if (k0 + n      > q) x0 = NEGINF;
        if (k0 + 16 + n > q) x1 = NEGINF;
        float t = fmaxf(x0, x1);                   // row max across 16-lane half
        t = fmaxf(t, __shfl_xor(t, 1, 32));
        t = fmaxf(t, __shfl_xor(t, 2, 32));
        t = fmaxf(t, __shfl_xor(t, 4, 32));
        t = fmaxf(t, __shfl_xor(t, 8, 32));
        const float nm    = fmaxf(rmax[r], t);
        const float alpha = __expf(rmax[r] - nm);
        const float p0 = __expf(x0 - nm);
        const float p1 = __expf(x1 - nm);
        float ps = p0 + p1;
        ps += __shfl_xor(ps, 1, 32);
        ps += __shfl_xor(ps, 2, 32);
        ps += __shfl_xor(ps, 4, 32);
        ps += __shfl_xor(ps, 8, 32);
        rsum[r] = rsum[r] * alpha + ps;
        rmax[r] = nm;
#pragma unroll
        for (int dt = 0; dt < 4; ++dt) o[dt][r] *= alpha;
        Ps[wave][(rsel + r) * PS_P + n]      = (_Float16)p0;
        Ps[wave][(rsel + r) * PS_P + 16 + n] = (_Float16)p1;
      }
      asm volatile("s_wait_dscnt 0" ::: "memory");  // P stores -> frag reload
      const v16h pa = frag_a(Ps[wave], PS_P, lane);
#pragma unroll
      for (int dt = 0; dt < 4; ++dt) {
        v16h bv = frag_b(Vt + dt * 16 * VT_P, VT_P, lane);
        o[dt] = wmma_f16(pa, bv, o[dt]);
      }
    }
    __syncthreads();
  }

  _Float16* Op = O + (rowbase + qbase) * D_MODEL + hoff;
#pragma unroll
  for (int dt = 0; dt < 4; ++dt)
#pragma unroll
    for (int r = 0; r < 8; ++r) {
      const float v = o[dt][r] / rsum[r];
      Op[(size_t)(rsel + r) * D_MODEL + dt * 16 + n] = (_Float16)v;
    }
}

// ---------------------------------------------------------------------------
// y = LayerNorm(x + r) * w + b ; fp32 out, optional f16 copy for next GEMM.
// One 256-thread block per row (D=1024 -> 4 elements/thread).
// ---------------------------------------------------------------------------
__global__ __launch_bounds__(256)
void ln_res_kernel(const float* __restrict__ X, const float* __restrict__ R,
                   const float* __restrict__ w, const float* __restrict__ bvec,
                   float* __restrict__ outF, _Float16* __restrict__ outH) {
  const int row = blockIdx.x;
  const int tid = threadIdx.x;
  const float* x  = X + (size_t)row * D_MODEL;
  const float* rr = R + (size_t)row * D_MODEL;
  __shared__ float red[8];
  float v[4];
  float s = 0.0f;
#pragma unroll
  for (int i = 0; i < 4; ++i) {
    const int c = tid + i * 256;
    v[i] = x[c] + rr[c];
    s += v[i];
  }
#pragma unroll
  for (int m = 1; m < 32; m <<= 1) s += __shfl_xor(s, m, 32);
  if ((tid & 31) == 0) red[tid >> 5] = s;
  __syncthreads();
  float tot = 0.0f;
#pragma unroll
  for (int i = 0; i < 8; ++i) tot += red[i];
  const float mean = tot * (1.0f / D_MODEL);
  __syncthreads();
  s = 0.0f;
#pragma unroll
  for (int i = 0; i < 4; ++i) { const float d = v[i] - mean; s += d * d; }
#pragma unroll
  for (int m = 1; m < 32; m <<= 1) s += __shfl_xor(s, m, 32);
  if ((tid & 31) == 0) red[tid >> 5] = s;
  __syncthreads();
  tot = 0.0f;
#pragma unroll
  for (int i = 0; i < 8; ++i) tot += red[i];
  const float inv = rsqrtf(tot * (1.0f / D_MODEL) + 1e-5f);
#pragma unroll
  for (int i = 0; i < 4; ++i) {
    const int c = tid + i * 256;
    const float y = (v[i] - mean) * inv * w[c] + bvec[c];
    outF[(size_t)row * D_MODEL + c] = y;
    if (outH) outH[(size_t)row * D_MODEL + c] = (_Float16)y;
  }
}

// ---------------------------------------------------------------------------
extern "C" void kernel_launch(void* const* d_in, const int* in_sizes, int n_in,
                              void* d_out, int out_size, void* d_ws, size_t ws_size,
                              hipStream_t stream) {
  const float* X     = (const float*)d_in[0];
  const float* Wq    = (const float*)d_in[1];
  const float* bq    = (const float*)d_in[2];
  const float* Wk    = (const float*)d_in[3];
  const float* bk    = (const float*)d_in[4];
  const float* Wv    = (const float*)d_in[5];
  const float* bv    = (const float*)d_in[6];
  const float* Wo    = (const float*)d_in[7];
  const float* bo    = (const float*)d_in[8];
  const float* ln1w  = (const float*)d_in[9];
  const float* ln1b  = (const float*)d_in[10];
  const float* Wup   = (const float*)d_in[11];
  const float* bup   = (const float*)d_in[12];
  const float* Wdown = (const float*)d_in[13];
  const float* bdown = (const float*)d_in[14];
  const float* ln2w  = (const float*)d_in[15];
  const float* ln2b  = (const float*)d_in[16];

  char* ws = (char*)d_ws;
  size_t off = 0;
  auto alloc = [&](size_t bytes) -> void* {
    void* p = ws + off;
    off += (bytes + 255) & ~(size_t)255;
    return p;
  };

  const size_t MD  = (size_t)MROWS * D_MODEL;
  const size_t DD  = (size_t)D_MODEL * D_MODEL;
  const size_t DFD = (size_t)DFF * D_MODEL;

  _Float16* Xh   = (_Float16*)alloc(MD * 2);
  _Float16* Wqh  = (_Float16*)alloc(DD * 2);
  _Float16* Wkh  = (_Float16*)alloc(DD * 2);
  _Float16* Wvh  = (_Float16*)alloc(DD * 2);
  _Float16* Woh  = (_Float16*)alloc(DD * 2);
  _Float16* Wuph = (_Float16*)alloc(DFD * 2);
  _Float16* Wdnh = (_Float16*)alloc(DFD * 2);
  _Float16* Qh   = (_Float16*)alloc(MD * 2);
  _Float16* Kh   = (_Float16*)alloc(MD * 2);
  _Float16* Vh   = (_Float16*)alloc(MD * 2);
  _Float16* AOh  = (_Float16*)alloc(MD * 2);
  float*    Proj = (float*)   alloc(MD * 4);
  float*    X1   = (float*)   alloc(MD * 4);
  _Float16* X1h  = (_Float16*)alloc(MD * 2);
  _Float16* Uph  = (_Float16*)alloc((size_t)MROWS * DFF * 2);
  float*    Down = (float*)   alloc(MD * 4);

  auto cvt = [&](const float* src, _Float16* dst, size_t n) {
    cvt_f16_kernel<<<(unsigned)((n + 255) / 256), 256, 0, stream>>>(src, dst, (int)n);
  };
  cvt(X,     Xh,   MD);
  cvt(Wq,    Wqh,  DD);
  cvt(Wk,    Wkh,  DD);
  cvt(Wv,    Wvh,  DD);
  cvt(Wo,    Woh,  DD);
  cvt(Wup,   Wuph, DFD);
  cvt(Wdown, Wdnh, DFD);

  dim3 gq(MROWS / GT, D_MODEL / GT);
  gemm_kernel<<<gq, 128, 0, stream>>>(Xh, Wqh, bq, nullptr, Qh, MROWS, D_MODEL, D_MODEL, 0);
  gemm_kernel<<<gq, 128, 0, stream>>>(Xh, Wkh, bk, nullptr, Kh, MROWS, D_MODEL, D_MODEL, 0);
  gemm_kernel<<<gq, 128, 0, stream>>>(Xh, Wvh, bv, nullptr, Vh, MROWS, D_MODEL, D_MODEL, 0);

  dim3 ga(SEQ / 64, NHEAD, BATCH);
  attn_kernel<<<ga, 128, 0, stream>>>(Qh, Kh, Vh, AOh);

  gemm_kernel<<<gq, 128, 0, stream>>>(AOh, Woh, bo, Proj, nullptr, MROWS, D_MODEL, D_MODEL, 0);
  ln_res_kernel<<<MROWS, 256, 0, stream>>>(X, Proj, ln1w, ln1b, X1, X1h);

  dim3 gu(MROWS / GT, DFF / GT);
  gemm_kernel<<<gu, 128, 0, stream>>>(X1h, Wuph, bup, nullptr, Uph, MROWS, DFF, D_MODEL, 1);
  gemm_kernel<<<gq, 128, 0, stream>>>(Uph, Wdnh, bdown, Down, nullptr, MROWS, D_MODEL, DFF, 0);
  ln_res_kernel<<<MROWS, 256, 0, stream>>>(X1, Down, ln2w, ln2b, (float*)d_out, nullptr);
}
